// DCRNN_Attack_45801531244823
// MI455X (gfx1250) — matrix operand order repaired and checked by reading, not compiled, not run on hardware
//
#include <hip/hip_runtime.h>

// ---------------------------------------------------------------------------
// DCRNN (K=1) fused kernel for gfx1250: bf16 WMMA (v_wmma_f32_16x16x32_bf16)
// One wave (32 lanes) processes a 16-row tile end-to-end.
// amdgpu_num_vgpr(256): cap register budget -> no s_set_vgpr_msb traffic,
// higher wave occupancy for this HBM-bound kernel.
// ---------------------------------------------------------------------------

typedef __attribute__((ext_vector_type(16))) __bf16 v16bf;
typedef __attribute__((ext_vector_type(8)))  __bf16 v8bf;
typedef __attribute__((ext_vector_type(4)))  __bf16 v4bf;
typedef __attribute__((ext_vector_type(8)))  float  v8f;

#define WMMA_BF16(A, B, Cc) \
  __builtin_amdgcn_wmma_f32_16x16x32_bf16(false, (A), false, (B), (short)0, (Cc), false, false)

// A fragment (16x32 bf16, row-major in LDS with arbitrary row stride):
// lane (ln,hi): elements 0..7  = row[ k0 + 8*hi + 0..7 ]
//               elements 8..15 = row[ k0 + 8*hi + 16..23 ]
__device__ __forceinline__ v16bf fragA(const __bf16* rowk0, int hi) {
  v8bf a0 = *(const v8bf*)(rowk0 + 8 * hi);
  v8bf a1 = *(const v8bf*)(rowk0 + 8 * hi + 16);
  v16bf r;
#pragma unroll
  for (int i = 0; i < 8; ++i) { r[i] = a0[i]; r[8 + i] = a1[i]; }
  return r;
}

// B fragment (32x16 bf16, weights stored transposed WT[n][k]):
// lane (ln,hi): elements 0..7  = WT[n][ k0 + 8*hi + 0..7 ]
//               elements 8..15 = WT[n][ k0 + 8*hi + 16..23 ]
__device__ __forceinline__ v16bf fragB(const __bf16* colk0, int hi) {
  v8bf b0 = *(const v8bf*)(colk0 + 8 * hi);
  v8bf b1 = *(const v8bf*)(colk0 + 8 * hi + 16);
  v16bf r;
#pragma unroll
  for (int i = 0; i < 8; ++i) { r[i] = b0[i]; r[8 + i] = b1[i]; }
  return r;
}

__device__ __forceinline__ float sigmoidf_(float x) { return 1.0f / (1.0f + __expf(-x)); }
__device__ __forceinline__ float tanhf_(float x) {
  float t = __expf(-2.0f * fabsf(x));
  float r = (1.0f - t) / (1.0f + t);
  return copysignf(r, x);
}

__attribute__((amdgpu_num_vgpr(256)))
__global__ __launch_bounds__(128) void dcrnn_fused_kernel(
    const float* __restrict__ x, const float* __restrict__ y, const float* __restrict__ H,
    const float* __restrict__ wz, const float* __restrict__ bz,
    const float* __restrict__ wr, const float* __restrict__ br,
    const float* __restrict__ wh, const float* __restrict__ bh,
    const float* __restrict__ lin_w, const float* __restrict__ lin_b,
    const float* __restrict__ lab_w1, const float* __restrict__ lab_b1,
    const float* __restrict__ lab_w2, const float* __restrict__ lab_b2,
    const float* __restrict__ comb_w, const float* __restrict__ comb_b,
    float* __restrict__ outSM, float* __restrict__ outH,
    int Nrows, int nChunks) {
  // ---- shared weights (bf16, transposed: WT[n][k]) ----
  __shared__ __bf16 sGate[3][32 * 160];   // (wz0+wz1)^T, (wr0+wr1)^T, (wh0+wh1)^T
  __shared__ __bf16 sLinT[16 * 32];       // lin_w^T   [n<10][k<32]  padded
  __shared__ __bf16 sLab1T[32 * 32];      // lab_w1^T  [n<32][k<10]  padded K->32
  __shared__ __bf16 sLab2T[16 * 32];      // lab_w2^T  [n<10][k<32]  padded
  __shared__ __bf16 sCombT[16 * 32];      // comb_w^T  [n<2][k<20]   padded
  __shared__ float sBz[32], sBr[32], sBh[32], sLab1b[32], sLinb[16], sLab2b[16], sCombb[16];
  // ---- per-wave scratch ----
  __shared__ __bf16 sXH[4][16 * 176];     // XH tile bf16, cols 0..127=x, 128..159=H
  __shared__ float  sHf[4][16 * 36];      // H tile f32 (for elementwise GRU math)
  __shared__ __bf16 sRHb[4][16 * 40];     // R*H tile (C-layout -> A-layout transpose)
  __shared__ __bf16 sHNb[4][16 * 40];     // relu(Hn) tile
  __shared__ __bf16 sYb[4][16 * 32];      // y tile padded K->32
  __shared__ __bf16 sY1b[4][16 * 40];     // relu(y@lab_w1+b1) tile
  __shared__ __bf16 sCBb[4][16 * 32];     // concat(h, y2) tile padded K->32

  const int tid = threadIdx.x;

  // ---------------- stage weights (block-cooperative, once) ----------------
  for (int idx = tid; idx < 3 * 5120; idx += 128) {
    int g = idx / 5120, r = idx - g * 5120;
    int n = r / 160, k = r - n * 160;
    const float* w = (g == 0) ? wz : (g == 1) ? wr : wh;
    sGate[g][n * 160 + k] = (__bf16)(w[k * 32 + n] + w[5120 + k * 32 + n]);
  }
  for (int idx = tid; idx < 512; idx += 128) {
    int n = idx >> 5, k = idx & 31;
    sLinT[idx]  = (n < 10) ? (__bf16)lin_w[k * 10 + n] : (__bf16)0.0f;
    sLab2T[idx] = (n < 10) ? (__bf16)lab_w2[k * 10 + n] : (__bf16)0.0f;
    sCombT[idx] = (n < 2 && k < 20) ? (__bf16)comb_w[k * 2 + n] : (__bf16)0.0f;
  }
  for (int idx = tid; idx < 1024; idx += 128) {
    int n = idx >> 5, k = idx & 31;
    sLab1T[idx] = (k < 10) ? (__bf16)lab_w1[k * 32 + n] : (__bf16)0.0f;
  }
  if (tid < 32) { sBz[tid] = bz[tid]; sBr[tid] = br[tid]; sBh[tid] = bh[tid]; sLab1b[tid] = lab_b1[tid]; }
  if (tid < 16) {
    sLinb[tid]  = (tid < 10) ? lin_b[tid]  : 0.0f;
    sLab2b[tid] = (tid < 10) ? lab_b2[tid] : 0.0f;
    sCombb[tid] = (tid < 2)  ? comb_b[tid] : 0.0f;
  }
  __syncthreads();

  const int lane = tid & 31, wid = tid >> 5;
  const int ln = lane & 15, hi = lane >> 4;
  const int waveGlobal = blockIdx.x * 4 + wid;
  const int totalWaves = gridDim.x * 4;

  __bf16* xh  = &sXH[wid][0];
  float*  hf  = &sHf[wid][0];
  __bf16* rhb = &sRHb[wid][0];
  __bf16* hnb = &sHNb[wid][0];
  __bf16* yb  = &sYb[wid][0];
  __bf16* y1b = &sY1b[wid][0];
  __bf16* cb  = &sCBb[wid][0];

  for (int ch = waveGlobal; ch < nChunks; ch += totalWaves) {
    const int row0 = ch * 16;

    // ---- stage x tile (coalesced float4 -> bf16 LDS) ----
#pragma unroll
    for (int i = 0; i < 16; ++i) {
      int idx4 = i * 32 + lane;               // [16][32] float4s
      int r = idx4 >> 5, c4 = idx4 & 31;
      int rg = row0 + r; if (rg >= Nrows) rg = Nrows - 1;
      float4 v = ((const float4*)x)[(size_t)rg * 32 + c4];
      v4bf b; b[0] = (__bf16)v.x; b[1] = (__bf16)v.y; b[2] = (__bf16)v.z; b[3] = (__bf16)v.w;
      *(v4bf*)(xh + r * 176 + c4 * 4) = b;
    }
    // ---- stage H tile: copy to outH, keep f32 + bf16 copies ----
#pragma unroll
    for (int i = 0; i < 4; ++i) {
      int idx4 = i * 32 + lane;               // [16][8] float4s
      int r = idx4 >> 3, c4 = idx4 & 7;
      int rg0 = row0 + r;
      int rg = (rg0 >= Nrows) ? Nrows - 1 : rg0;
      float4 v = ((const float4*)H)[(size_t)rg * 8 + c4];
      if (rg0 < Nrows) ((float4*)outH)[(size_t)rg0 * 8 + c4] = v;
      v4bf b; b[0] = (__bf16)v.x; b[1] = (__bf16)v.y; b[2] = (__bf16)v.z; b[3] = (__bf16)v.w;
      *(v4bf*)(xh + r * 176 + 128 + c4 * 4) = b;
      *(float4*)(hf + r * 36 + c4 * 4) = v;
    }
    // ---- stage y tile (pad cols 10..31 with zero) ----
#pragma unroll
    for (int i = 0; i < 16; ++i) {
      int idx = i * 32 + lane;
      int r = idx >> 5, c = idx & 31;
      int rg = row0 + r; if (rg >= Nrows) rg = Nrows - 1;
      float v = (c < 10) ? y[(size_t)rg * 10 + c] : 0.0f;
      yb[r * 32 + c] = (__bf16)v;
    }
    // ---- zero concat buffer (overwritten partially later) ----
    {
      v8bf z;
#pragma unroll
      for (int i = 0; i < 8; ++i) z[i] = (__bf16)0.0f;
      *(v8bf*)(cb + ln * 32 + hi * 16) = z;
      *(v8bf*)(cb + ln * 32 + hi * 16 + 8) = z;
    }

    // ================= gate GEMMs: Z, R, Ht(x-part) =================
    v8f accZ0 = {}, accZ1 = {}, accR0 = {}, accR1 = {}, accH0 = {}, accH1 = {};
#pragma unroll
    for (int kk = 0; kk < 5; ++kk) {
      const int k0 = kk * 32;
      v16bf a = fragA(xh + ln * 176 + k0, hi);
      v16bf b0, b1;
      b0 = fragB(&sGate[0][ln * 160 + k0], hi);
      b1 = fragB(&sGate[0][(16 + ln) * 160 + k0], hi);
      accZ0 = WMMA_BF16(a, b0, accZ0);
      accZ1 = WMMA_BF16(a, b1, accZ1);
      b0 = fragB(&sGate[1][ln * 160 + k0], hi);
      b1 = fragB(&sGate[1][(16 + ln) * 160 + k0], hi);
      accR0 = WMMA_BF16(a, b0, accR0);
      accR1 = WMMA_BF16(a, b1, accR1);
      if (kk < 4) {  // Ht: rows 0..127 multiply x only
        b0 = fragB(&sGate[2][ln * 160 + k0], hi);
        b1 = fragB(&sGate[2][(16 + ln) * 160 + k0], hi);
        accH0 = WMMA_BF16(a, b0, accH0);
        accH1 = WMMA_BF16(a, b1, accH1);
      }
    }

    // ---- R = sigmoid(.), RH = R*H ; transpose through LDS ----
    float hC0[8], hC1[8];
#pragma unroll
    for (int j = 0; j < 8; ++j) {
      hC0[j] = hf[(j + 8 * hi) * 36 + ln];
      hC1[j] = hf[(j + 8 * hi) * 36 + 16 + ln];
      float r0 = sigmoidf_(accR0[j] + sBr[ln]);
      float r1 = sigmoidf_(accR1[j] + sBr[16 + ln]);
      rhb[(j + 8 * hi) * 40 + ln]      = (__bf16)(r0 * hC0[j]);
      rhb[(j + 8 * hi) * 40 + 16 + ln] = (__bf16)(r1 * hC1[j]);
    }
    asm volatile("" ::: "memory");  // LDS same-wave ops are in-order; block compile-time reordering

    // ---- Ht tail: (R*H) @ Wh[128..159,:] ----
    {
      v16bf a = fragA(rhb + ln * 40, hi);
      v16bf b0 = fragB(&sGate[2][ln * 160 + 128], hi);
      v16bf b1 = fragB(&sGate[2][(16 + ln) * 160 + 128], hi);
      accH0 = WMMA_BF16(a, b0, accH0);
      accH1 = WMMA_BF16(a, b1, accH1);
    }

    // ---- GRU combine, relu(Hn) -> LDS ----
#pragma unroll
    for (int j = 0; j < 8; ++j) {
      float z0 = sigmoidf_(accZ0[j] + sBz[ln]);
      float z1 = sigmoidf_(accZ1[j] + sBz[16 + ln]);
      float t0 = tanhf_(accH0[j] + sBh[ln]);
      float t1 = tanhf_(accH1[j] + sBh[16 + ln]);
      float hn0 = z0 * hC0[j] + (1.0f - z0) * t0;
      float hn1 = z1 * hC1[j] + (1.0f - z1) * t1;
      hnb[(j + 8 * hi) * 40 + ln]      = (__bf16)fmaxf(hn0, 0.0f);
      hnb[(j + 8 * hi) * 40 + 16 + ln] = (__bf16)fmaxf(hn1, 0.0f);
    }
    asm volatile("" ::: "memory");

    // ================= head (all WMMA, padded shapes) =================
    // h = relu(Hn) @ lin_w  (K=32, N=10 padded to 16)
    v8f accL = {};
    {
      v16bf a = fragA(hnb + ln * 40, hi);
      v16bf b = fragB(&sLinT[ln * 32], hi);
      accL = WMMA_BF16(a, b, accL);
    }
    // y1 = relu(y @ lab_w1 + b1)  (K=10 padded to 32, N=32)
    v8f accY10 = {}, accY11 = {};
    {
      v16bf a = fragA(yb + ln * 32, hi);
      v16bf b0 = fragB(&sLab1T[ln * 32], hi);
      v16bf b1 = fragB(&sLab1T[(16 + ln) * 32], hi);
      accY10 = WMMA_BF16(a, b0, accY10);
      accY11 = WMMA_BF16(a, b1, accY11);
    }
#pragma unroll
    for (int j = 0; j < 8; ++j) {
      y1b[(j + 8 * hi) * 40 + ln]      = (__bf16)fmaxf(accY10[j] + sLab1b[ln], 0.0f);
      y1b[(j + 8 * hi) * 40 + 16 + ln] = (__bf16)fmaxf(accY11[j] + sLab1b[16 + ln], 0.0f);
    }
    asm volatile("" ::: "memory");
    // y2 = relu(y1 @ lab_w2 + b2)  (K=32, N=10 padded)
    v8f accY2 = {};
    {
      v16bf a = fragA(y1b + ln * 40, hi);
      v16bf b = fragB(&sLab2T[ln * 32], hi);
      accY2 = WMMA_BF16(a, b, accY2);
    }
    // concat(h, y2) -> cb (cols 0..9 = h, 10..19 = y2, rest stays zero)
    if (ln < 10) {
#pragma unroll
      for (int j = 0; j < 8; ++j) {
        cb[(j + 8 * hi) * 32 + ln]      = (__bf16)(accL[j] + sLinb[ln]);
        cb[(j + 8 * hi) * 32 + 10 + ln] = (__bf16)fmaxf(accY2[j] + sLab2b[ln], 0.0f);
      }
    }
    asm volatile("" ::: "memory");
    // logits = concat @ comb_w  (K=20 padded to 32, N=2 padded)
    v8f accC = {};
    {
      v16bf a = fragA(cb + ln * 32, hi);
      v16bf b = fragB(&sCombT[ln * 32], hi);
      accC = WMMA_BF16(a, b, accC);
    }

    // ---- 2-wide softmax via lane swap; store [N,2] ----
#pragma unroll
    for (int j = 0; j < 8; ++j) {
      float own = accC[j] + sCombb[ln];
      float oth = __shfl_xor(own, 1);
      float p = 1.0f / (1.0f + __expf(oth - own));
      int rg = row0 + j + 8 * hi;
      if (ln < 2 && rg < Nrows) outSM[(size_t)rg * 2 + ln] = p;
    }
  }
}

extern "C" void kernel_launch(void* const* d_in, const int* in_sizes, int n_in,
                              void* d_out, int out_size, void* d_ws, size_t ws_size,
                              hipStream_t stream) {
  const float* x      = (const float*)d_in[0];
  const float* y      = (const float*)d_in[1];
  const float* H      = (const float*)d_in[2];
  // d_in[3] edge_index, d_in[4] edge_weight: unused (K=1 DConv has no propagation)
  const float* wz     = (const float*)d_in[5];
  const float* bz     = (const float*)d_in[6];
  const float* wr     = (const float*)d_in[7];
  const float* br     = (const float*)d_in[8];
  const float* wh     = (const float*)d_in[9];
  const float* bh     = (const float*)d_in[10];
  const float* lin_w  = (const float*)d_in[11];
  const float* lin_b  = (const float*)d_in[12];
  const float* lab_w1 = (const float*)d_in[13];
  const float* lab_b1 = (const float*)d_in[14];
  const float* lab_w2 = (const float*)d_in[15];
  const float* lab_b2 = (const float*)d_in[16];
  const float* comb_w = (const float*)d_in[17];
  const float* comb_b = (const float*)d_in[18];

  const int Nrows   = in_sizes[0] / 128;
  const int nChunks = (Nrows + 15) / 16;

  float* outSM = (float*)d_out;
  float* outH  = outSM + (size_t)Nrows * 2;

  int blocks = (nChunks + 3) / 4;       // 4 waves per block, 1 tile per wave
  if (blocks < 1) blocks = 1;
  if (blocks > 4096) blocks = 4096;     // grid-stride over tiles beyond this

  dcrnn_fused_kernel<<<dim3(blocks), dim3(128), 0, stream>>>(
      x, y, H, wz, bz, wr, br, wh, bh, lin_w, lin_b,
      lab_w1, lab_b1, lab_w2, lab_b2, comb_w, comb_b,
      outSM, outH, Nrows, nChunks);
}
